// Model_69037304315982
// MI455X (gfx1250) — compile-verified, matrix operand order checked by reading, not run on hardware
//
#include <hip/hip_runtime.h>
#include <hip/hip_bf16.h>
#include <math.h>
#include <stdint.h>

// ---------------------------------------------------------------------------
// Problem constants (match reference)
// ---------------------------------------------------------------------------
#define VV  10000
#define HH  768
#define BB  8
#define SS  512
#define LL  8
#define MM  (BB * SS)          // 4096 rows of activations
#define EPS 1e-5f

// GEMM layout/epilogue flags (compile-time template parameter)
#define GF_BT     1   // B is [N,K] (K-contiguous per output column)
#define GF_BF32   2   // B source is f32 (convert to bf16 while staging)
#define GF_RELU   4
#define GF_CAUSAL 8   // scores mask: -inf where col(k) > row(q)

typedef __attribute__((ext_vector_type(16))) __bf16        v16bf;
typedef __attribute__((ext_vector_type(8)))  __bf16        v8bf;
typedef __attribute__((ext_vector_type(8)))  float         v8f;
typedef __attribute__((ext_vector_type(4)))  unsigned int  v4u;
typedef __attribute__((ext_vector_type(4)))  int           v4i;
typedef __attribute__((ext_vector_type(8)))  int           v8i;

#if defined(__has_builtin)
#if __has_builtin(__builtin_amdgcn_tensor_load_to_lds) && \
    __has_builtin(__builtin_amdgcn_s_wait_tensorcnt)
#define USE_TDM 1
#endif
#endif
#ifndef USE_TDM
#define USE_TDM 0
#endif

static __device__ __forceinline__ unsigned short f2bf(float v) {
    __hip_bfloat16 h = __float2bfloat16(v);
    return *reinterpret_cast<unsigned short*>(&h);
}

// ---------------------------------------------------------------------------
// Generic bf16 WMMA GEMM:  C[z] = alpha * (A[z] @ B[z]) + bias (+mask/relu/res)
//   A  : [M,K] bf16 row-major, per-batch stride sA (elements)
//   B  : FLAGS&GF_BT ? [N,K] : [K,N]; f32 or bf16 per GF_BF32; stride sB
//   C  : f32 (Cf) and/or bf16 (Cb), [M,N], stride sC; residual Res uses sC too
//
// 256 threads = 8 waves (4x2), macro tile 128x128, wave tile 32x64:
// 2x4 v_wmma_f32_16x16x32_bf16 per 32-deep K step.
// Double-buffered LDS; A tile is DMA'd by the Tensor Data Mover
// (tensor_load_to_lds + s_wait_tensorcnt) when the toolchain exposes it,
// B tile is double-buffered through registers.  M%128==0, K%32==0 required.
// ---------------------------------------------------------------------------
template<int FLAGS>
__launch_bounds__(256)
__global__ void gemm_bf16_wmma(const __hip_bfloat16* __restrict__ A,
                               const void* __restrict__ Bp,
                               const float* __restrict__ bias,
                               const float* __restrict__ Res,
                               float* __restrict__ Cf,
                               __hip_bfloat16* __restrict__ Cb,
                               int M, int N, int K,
                               long long sA, long long sB, long long sC,
                               float alpha)
{
    constexpr bool BT     = (FLAGS & GF_BT)     != 0;
    constexpr bool BF32   = (FLAGS & GF_BF32)   != 0;
    constexpr bool RELU   = (FLAGS & GF_RELU)   != 0;
    constexpr bool CAUSAL = (FLAGS & GF_CAUSAL) != 0;

    __shared__ __align__(16) unsigned short At[2][128 * 32];   // [m][k]
    __shared__ __align__(16) unsigned short Bt[2][128 * 32];   // [n][k]

    const int tid  = threadIdx.x;
    const int lane = tid & 31;
    const int wave = tid >> 5;
    const int z    = blockIdx.z;
    const int m0   = blockIdx.y * 128;
    const int n0   = blockIdx.x * 128;

    A += (long long)z * sA;
    const float*          Bf = (const float*)Bp          + (long long)z * sB;
    const __hip_bfloat16* Bh = (const __hip_bfloat16*)Bp + (long long)z * sB;

    const int mW = (wave >> 1) * 32;       // 4 waves along M (32-row strips)
    const int nW = (wave & 1) * 64;        // 2 waves along N (64-col strips)
    const int lrow    = lane & 15;
    const int halfSel = (lane >> 4) << 3;  // 0 (lanes 0-15) or 8 (lanes 16-31)
    const int rowAdd  = (lane >> 4) << 3;  // C/D layout: +8 rows for hi half

    v8f acc[2][4];
#pragma unroll
    for (int i = 0; i < 2; ++i)
#pragma unroll
        for (int j = 0; j < 4; ++j)
#pragma unroll
            for (int e = 0; e < 8; ++e) acc[i][j][e] = 0.0f;

    // ------ double-buffer staging registers ------
    uint4  rA[2];
    uint4  rBv[2];
    float4 rBf[2][2];
    float  rBt[2][8];

#if USE_TDM
    // One wave issues a TDM descriptor for the whole 128x32 bf16 A tile.
    auto tdm_load_A = [&](int kk, int buf) {
        if (wave != 0) return;
        const unsigned long long ga =
            (unsigned long long)(uintptr_t)(A + (long long)m0 * K + kk);
        const unsigned lds_off = (unsigned)(uintptr_t)(&At[buf][0]);
        v4u g0;
        g0[0] = 1u;                                     // count=1 (valid D#)
        g0[1] = lds_off;                                // lds_addr
        g0[2] = (unsigned)(ga & 0xffffffffu);           // global_addr[31:0]
        g0[3] = (unsigned)((ga >> 32) & 0x01ffffffu)    // global_addr[56:32]
              | (2u << 30);                             // type=2 ("image")
        v8i g1;
        g1[0] = 0x00010000;                             // data_size=1 -> 2 bytes
        g1[1] = (int)(((unsigned)K & 0xffffu) << 16);   // tensor_dim0[15:0]
        g1[2] = (int)((((unsigned)K >> 16) & 0xffffu)   // tensor_dim0[31:16]
              | (((unsigned)M & 0xffffu) << 16));       // tensor_dim1[15:0]
        g1[3] = (int)((((unsigned)M >> 16) & 0xffffu)   // tensor_dim1[31:16]
              | (32u << 16));                           // tile_dim0 = 32
        g1[4] = 128;                                    // tile_dim1 = 128
        g1[5] = K;                                      // tensor_dim0_stride
        g1[6] = 0;
        g1[7] = 0;
        v4i z4 = {0, 0, 0, 0};
#if defined(__clang_major__) && __clang_major__ >= 23
        v8i z8 = {0, 0, 0, 0, 0, 0, 0, 0};
        __builtin_amdgcn_tensor_load_to_lds(g0, g1, z4, z4, z8, 0);
#else
        __builtin_amdgcn_tensor_load_to_lds(g0, g1, z4, z4, 0);
#endif
    };
#else
    auto load_A = [&](int kk) {
#pragma unroll
        for (int it = 0; it < 2; ++it) {
            const int slot = tid + it * 256;
            const int row  = slot >> 2;
            const int seg  = (slot & 3) << 3;
            rA[it] = *(const uint4*)(A + (long long)(m0 + row) * K + kk + seg);
        }
    };
    auto store_A = [&](int buf) {
#pragma unroll
        for (int it = 0; it < 2; ++it) {
            const int slot = tid + it * 256;
            const int row  = slot >> 2;
            const int seg  = (slot & 3) << 3;
            *(uint4*)(&At[buf][row * 32 + seg]) = rA[it];
        }
    };
#endif

    auto load_B = [&](int kk) {
        if constexpr (BT) {                // B[N,K]: rows match LDS layout
#pragma unroll
            for (int it = 0; it < 2; ++it) {
                const int slot = tid + it * 256;
                const int row  = slot >> 2;
                const int seg  = (slot & 3) << 3;
                const int gn   = n0 + row;
                if constexpr (BF32) {
                    if (gn < N) {
                        const float* p = Bf + (long long)gn * K + kk + seg;
                        rBf[it][0] = *(const float4*)(p);
                        rBf[it][1] = *(const float4*)(p + 4);
                    } else {
                        rBf[it][0] = make_float4(0.f, 0.f, 0.f, 0.f);
                        rBf[it][1] = make_float4(0.f, 0.f, 0.f, 0.f);
                    }
                } else {
                    rBv[it] = (gn < N)
                        ? *(const uint4*)(Bh + (long long)gn * K + kk + seg)
                        : make_uint4(0u, 0u, 0u, 0u);
                }
            }
        } else {                           // B[K,N]: gather rows, transpose later
#pragma unroll
            for (int it = 0; it < 2; ++it) {
                const int slot = tid + it * 256;
                const int kr   = slot >> 4;
                const int nc   = (slot & 15) << 3;
                if constexpr (BF32) {
                    const float* p = Bf + (long long)(kk + kr) * N + n0 + nc;
#pragma unroll
                    for (int j = 0; j < 8; ++j)
                        rBt[it][j] = (n0 + nc + j < N) ? p[j] : 0.0f;
                } else {
                    const __hip_bfloat16* p = Bh + (long long)(kk + kr) * N + n0 + nc;
#pragma unroll
                    for (int j = 0; j < 8; ++j)
                        rBt[it][j] = (n0 + nc + j < N) ? __bfloat162float(p[j]) : 0.0f;
                }
            }
        }
    };

    auto store_B = [&](int buf) {
        if constexpr (BT) {
#pragma unroll
            for (int it = 0; it < 2; ++it) {
                const int slot = tid + it * 256;
                const int row  = slot >> 2;
                const int seg  = (slot & 3) << 3;
                if constexpr (BF32) {
                    unsigned short tmp[8] = {
                        f2bf(rBf[it][0].x), f2bf(rBf[it][0].y),
                        f2bf(rBf[it][0].z), f2bf(rBf[it][0].w),
                        f2bf(rBf[it][1].x), f2bf(rBf[it][1].y),
                        f2bf(rBf[it][1].z), f2bf(rBf[it][1].w)};
                    *(uint4*)(&Bt[buf][row * 32 + seg]) = *(const uint4*)tmp;
                } else {
                    *(uint4*)(&Bt[buf][row * 32 + seg]) = rBv[it];
                }
            }
        } else {
#pragma unroll
            for (int it = 0; it < 2; ++it) {
                const int slot = tid + it * 256;
                const int kr   = slot >> 4;
                const int nc   = (slot & 15) << 3;
#pragma unroll
                for (int j = 0; j < 8; ++j)
                    Bt[buf][(nc + j) * 32 + kr] = f2bf(rBt[it][j]);
            }
        }
    };

    auto compute = [&](int buf) {
        v16bf afr[2], bfr[4];
#pragma unroll
        for (int mf = 0; mf < 2; ++mf) {
            const unsigned short* p = &At[buf][(mW + mf * 16 + lrow) * 32 + halfSel];
            const v8bf lo = *(const v8bf*)(p);        // K 0..7   (or 8..15)
            const v8bf hi = *(const v8bf*)(p + 16);   // K 16..23 (or 24..31)
#pragma unroll
            for (int e = 0; e < 8; ++e) { afr[mf][e] = lo[e]; afr[mf][8 + e] = hi[e]; }
        }
#pragma unroll
        for (int nf = 0; nf < 4; ++nf) {
            const unsigned short* p = &Bt[buf][(nW + nf * 16 + lrow) * 32 + halfSel];
            const v8bf lo = *(const v8bf*)(p);
            const v8bf hi = *(const v8bf*)(p + 16);
#pragma unroll
            for (int e = 0; e < 8; ++e) { bfr[nf][e] = lo[e]; bfr[nf][8 + e] = hi[e]; }
        }
#pragma unroll
        for (int mf = 0; mf < 2; ++mf)
#pragma unroll
            for (int nf = 0; nf < 4; ++nf)
                acc[mf][nf] = __builtin_amdgcn_wmma_f32_16x16x32_bf16(
                    false, afr[mf], false, bfr[nf], (short)0, acc[mf][nf],
                    false, false);
    };

    // ------ prologue: stage K-step 0 into buffer 0 ------
    const int ksteps = K >> 5;
#if USE_TDM
    tdm_load_A(0, 0);
#else
    load_A(0);
#endif
    load_B(0);
#if !USE_TDM
    store_A(0);
#endif
    store_B(0);
#if USE_TDM
    if (wave == 0) __builtin_amdgcn_s_wait_tensorcnt(0);
#endif
    __syncthreads();

    // ------ main loop: one barrier per K-step, loads overlap WMMA ------
    int buf = 0;
    for (int ks = 0; ks < ksteps; ++ks) {
        const int kk = ks << 5;
        const bool have_next = (ks + 1 < ksteps);
        if (have_next) {
#if USE_TDM
            tdm_load_A(kk + 32, buf ^ 1);       // DMA next A tile (async)
#else
            load_A(kk + 32);
#endif
            load_B(kk + 32);                    // issue global loads early
            __builtin_prefetch(A + (long long)(m0 + (tid >> 1)) * K + kk + 64, 0, 1);
        }
        compute(buf);                           // 8x v_wmma on current buffer
        if (have_next) {
#if !USE_TDM
            store_A(buf ^ 1);
#endif
            store_B(buf ^ 1);
#if USE_TDM
            if (wave == 0) __builtin_amdgcn_s_wait_tensorcnt(0);
#endif
        }
        __syncthreads();
        buf ^= 1;
    }

    // ------ epilogue: scale, bias, mask, relu, residual, store ------
#pragma unroll
    for (int mf = 0; mf < 2; ++mf) {
#pragma unroll
        for (int nf = 0; nf < 4; ++nf) {
            const int gn = n0 + nW + nf * 16 + lrow;
            if (gn >= N) continue;
#pragma unroll
            for (int e = 0; e < 8; ++e) {
                const int gm = m0 + mW + mf * 16 + e + rowAdd;
                float val = acc[mf][nf][e] * alpha;
                if (bias) val += bias[gn];
                if constexpr (CAUSAL) { if (gn > gm) val = -1e30f; }
                if constexpr (RELU)   val = fmaxf(val, 0.0f);
                const long long idx = (long long)z * sC + (long long)gm * N + gn;
                if (Res) val += Res[idx];
                if (Cf) Cf[idx] = val;
                if (Cb) Cb[idx] = __float2bfloat16(val);
            }
        }
    }
}

// ---------------------------------------------------------------------------
// Embedding gather + sinusoidal positional encoding -> z (f32 + bf16)
// ---------------------------------------------------------------------------
__global__ void embed_kernel(const int* __restrict__ x,
                             const float* __restrict__ emb,
                             float* __restrict__ zf,
                             __hip_bfloat16* __restrict__ zb)
{
    const int row = blockIdx.x;          // 0..MM-1
    const int s   = row & (SS - 1);
    const int tok = x[row];
    const float* e = emb + (long long)tok * HH;
    const float negln = -logf(10000.0f) / (float)HH;
    for (int h = threadIdx.x; h < HH; h += blockDim.x) {
        const int   i2  = (h >> 1) << 1;
        const float ang = (float)s * __expf((float)i2 * negln);
        const float pe  = (h & 1) ? __cosf(ang) : __sinf(ang);
        const float v   = e[h] + pe;
        zf[(long long)row * HH + h] = v;
        zb[(long long)row * HH + h] = __float2bfloat16(v);
    }
}

// ---------------------------------------------------------------------------
// LayerNorm over H=768; one 256-thread block per row (3 elems/thread)
// ---------------------------------------------------------------------------
__launch_bounds__(256)
__global__ void layernorm_kernel(const float* __restrict__ X,
                                 const float* __restrict__ g,
                                 const float* __restrict__ b,
                                 float* __restrict__ Of,
                                 __hip_bfloat16* __restrict__ Ob)
{
    __shared__ float red[256];
    const int row = blockIdx.x;
    const float* xr = X + (long long)row * HH;
    float v[3], s = 0.0f, s2 = 0.0f;
#pragma unroll
    for (int i = 0; i < 3; ++i) {
        v[i] = xr[threadIdx.x + i * 256];
        s += v[i]; s2 += v[i] * v[i];
    }
    red[threadIdx.x] = s; __syncthreads();
    for (int o = 128; o > 0; o >>= 1) {
        if (threadIdx.x < o) red[threadIdx.x] += red[threadIdx.x + o];
        __syncthreads();
    }
    const float mean = red[0] / (float)HH; __syncthreads();
    red[threadIdx.x] = s2; __syncthreads();
    for (int o = 128; o > 0; o >>= 1) {
        if (threadIdx.x < o) red[threadIdx.x] += red[threadIdx.x + o];
        __syncthreads();
    }
    const float var = red[0] / (float)HH - mean * mean;
    const float inv = rsqrtf(var + EPS);
#pragma unroll
    for (int i = 0; i < 3; ++i) {
        const int h = threadIdx.x + i * 256;
        const float o = (v[i] - mean) * inv * g[h] + b[h];
        if (Of) Of[(long long)row * HH + h] = o;
        Ob[(long long)row * HH + h] = __float2bfloat16(o);
    }
}

// ---------------------------------------------------------------------------
// Faithful softmax over the QUERY axis (reference's dim=1): one block per
// (b,k) column; 512 strided entries, 2 per thread.
// ---------------------------------------------------------------------------
__launch_bounds__(256)
__global__ void softmax_q_kernel(const float* __restrict__ scores,
                                 __hip_bfloat16* __restrict__ atten)
{
    __shared__ float red[256];
    const int col = blockIdx.x;          // b*S + k
    const int b   = col >> 9;
    const int k   = col & (SS - 1);
    const float* base = scores + (long long)b * SS * SS + k;
    const float a0 = base[(long long)threadIdx.x * SS];
    const float a1 = base[(long long)(threadIdx.x + 256) * SS];

    red[threadIdx.x] = fmaxf(a0, a1); __syncthreads();
    for (int o = 128; o > 0; o >>= 1) {
        if (threadIdx.x < o) red[threadIdx.x] = fmaxf(red[threadIdx.x], red[threadIdx.x + o]);
        __syncthreads();
    }
    const float mx = red[0]; __syncthreads();

    const float e0 = __expf(a0 - mx), e1 = __expf(a1 - mx);
    red[threadIdx.x] = e0 + e1; __syncthreads();
    for (int o = 128; o > 0; o >>= 1) {
        if (threadIdx.x < o) red[threadIdx.x] += red[threadIdx.x + o];
        __syncthreads();
    }
    const float inv = 1.0f / red[0];

    __hip_bfloat16* ob = atten + (long long)b * SS * SS + k;
    ob[(long long)threadIdx.x * SS]         = __float2bfloat16(e0 * inv);
    ob[(long long)(threadIdx.x + 256) * SS] = __float2bfloat16(e1 * inv);
}

// ---------------------------------------------------------------------------
// Final softmax over V=10000, in place on d_out; one block per row
// ---------------------------------------------------------------------------
__launch_bounds__(256)
__global__ void softmax_v_kernel(float* __restrict__ logits)
{
    __shared__ float red[256];
    float* p = logits + (long long)blockIdx.x * VV;
    float mx = -1e30f;
    for (int i = threadIdx.x; i < VV; i += 256) mx = fmaxf(mx, p[i]);
    red[threadIdx.x] = mx; __syncthreads();
    for (int o = 128; o > 0; o >>= 1) {
        if (threadIdx.x < o) red[threadIdx.x] = fmaxf(red[threadIdx.x], red[threadIdx.x + o]);
        __syncthreads();
    }
    mx = red[0]; __syncthreads();
    float s = 0.0f;
    for (int i = threadIdx.x; i < VV; i += 256) s += __expf(p[i] - mx);
    red[threadIdx.x] = s; __syncthreads();
    for (int o = 128; o > 0; o >>= 1) {
        if (threadIdx.x < o) red[threadIdx.x] += red[threadIdx.x + o];
        __syncthreads();
    }
    const float inv = 1.0f / red[0];
    for (int i = threadIdx.x; i < VV; i += 256) p[i] = __expf(p[i] - mx) * inv;
}

// ---------------------------------------------------------------------------
// Host orchestration
// ---------------------------------------------------------------------------
extern "C" void kernel_launch(void* const* d_in, const int* in_sizes, int n_in,
                              void* d_out, int out_size, void* d_ws, size_t ws_size,
                              hipStream_t stream)
{
    (void)in_sizes; (void)n_in; (void)out_size; (void)ws_size;

    const int*   x    = (const int*)d_in[0];
    const float* emb  = (const float*)d_in[2];
    const float* Wq   = (const float*)d_in[3];
    const float* bq   = (const float*)d_in[4];
    const float* Wk   = (const float*)d_in[5];
    const float* bk   = (const float*)d_in[6];
    const float* Wv   = (const float*)d_in[7];
    const float* bv   = (const float*)d_in[8];
    const float* W1   = (const float*)d_in[9];
    const float* b1   = (const float*)d_in[10];
    const float* W2   = (const float*)d_in[11];
    const float* b2   = (const float*)d_in[12];
    const float* g1   = (const float*)d_in[13];
    const float* be1  = (const float*)d_in[14];
    const float* g2   = (const float*)d_in[15];
    const float* be2  = (const float*)d_in[16];
    const float* fcw  = (const float*)d_in[17];
    const float* fcb  = (const float*)d_in[18];
    float* out = (float*)d_out;

    // workspace layout
    char* ws = (char*)d_ws;
    size_t off = 0;
    auto alloc = [&](size_t bytes) -> char* {
        char* p = ws + off;
        off += (bytes + 255) & ~(size_t)255;
        return p;
    };
    float*          z_f  = (float*)alloc((size_t)MM * HH * 4);
    __hip_bfloat16* z_b  = (__hip_bfloat16*)alloc((size_t)MM * HH * 2);
    __hip_bfloat16* q_b  = (__hip_bfloat16*)alloc((size_t)MM * HH * 2);
    __hip_bfloat16* k_b  = (__hip_bfloat16*)alloc((size_t)MM * HH * 2);
    __hip_bfloat16* v_b  = (__hip_bfloat16*)alloc((size_t)MM * HH * 2);
    float*          sc_f = (float*)alloc((size_t)BB * SS * SS * 4);
    __hip_bfloat16* at_b = (__hip_bfloat16*)alloc((size_t)BB * SS * SS * 2);
    float*          t0_f = (float*)alloc((size_t)MM * HH * 4);
    float*          o1_f = (float*)alloc((size_t)MM * HH * 4);
    __hip_bfloat16* o1_b = (__hip_bfloat16*)alloc((size_t)MM * HH * 2);
    __hip_bfloat16* h1_b = (__hip_bfloat16*)alloc((size_t)MM * HH * 2);
    __hip_bfloat16* zf_b = (__hip_bfloat16*)alloc((size_t)MM * HH * 2);

    const dim3 blk(256);
    auto gemm = [&](const __hip_bfloat16* A, const void* B, const float* bias,
                    const float* Res, float* Cf, __hip_bfloat16* Cb,
                    int M, int N, int K, int batch,
                    long long sA, long long sB, long long sC,
                    float alpha, int flags) {
        dim3 grid((N + 127) / 128, M / 128, batch);
#define LAUNCH_GEMM(F)                                                        \
        gemm_bf16_wmma<F><<<grid, blk, 0, stream>>>(A, B, bias, Res, Cf, Cb,  \
                                                    M, N, K, sA, sB, sC, alpha)
        switch (flags) {
            case 0:                    LAUNCH_GEMM(0); break;
            case GF_BF32:              LAUNCH_GEMM(GF_BF32); break;
            case GF_BF32 | GF_RELU:    LAUNCH_GEMM(GF_BF32 | GF_RELU); break;
            case GF_BT | GF_CAUSAL:    LAUNCH_GEMM(GF_BT | GF_CAUSAL); break;
            case GF_BT | GF_BF32:      LAUNCH_GEMM(GF_BT | GF_BF32); break;
            default: break;
        }
#undef LAUNCH_GEMM
    };

    embed_kernel<<<MM, blk, 0, stream>>>(x, emb, z_f, z_b);

    const float scale = 1.0f / sqrtf((float)HH);
    for (int l = 0; l < LL; ++l) {
        const float* wq = Wq + (size_t)l * HH * HH;  const float* bql = bq + (size_t)l * HH;
        const float* wk = Wk + (size_t)l * HH * HH;  const float* bkl = bk + (size_t)l * HH;
        const float* wv = Wv + (size_t)l * HH * HH;  const float* bvl = bv + (size_t)l * HH;
        const float* w1 = W1 + (size_t)l * HH * HH;  const float* b1l = b1 + (size_t)l * HH;
        const float* w2 = W2 + (size_t)l * HH * HH;  const float* b2l = b2 + (size_t)l * HH;

        // q/k/v = z @ W + b   (weights f32 [K,N], converted while staging)
        gemm(z_b, wq, bql, nullptr, nullptr, q_b, MM, HH, HH, 1, 0, 0, 0, 1.0f, GF_BF32);
        gemm(z_b, wk, bkl, nullptr, nullptr, k_b, MM, HH, HH, 1, 0, 0, 0, 1.0f, GF_BF32);
        gemm(z_b, wv, bvl, nullptr, nullptr, v_b, MM, HH, HH, 1, 0, 0, 0, 1.0f, GF_BF32);

        // scores[b] = scale * q[b] @ k[b]^T, causal mask baked in (batched over z)
        gemm(q_b, k_b, nullptr, nullptr, sc_f, nullptr, SS, SS, HH, BB,
             (long long)SS * HH, (long long)SS * HH, (long long)SS * SS,
             scale, GF_BT | GF_CAUSAL);

        // faithful softmax over query axis
        softmax_q_kernel<<<BB * SS, blk, 0, stream>>>(sc_f, at_b);

        // out[b] = atten[b] @ v[b] + z  (residual fused)
        gemm(at_b, v_b, nullptr, z_f, t0_f, nullptr, SS, HH, SS, BB,
             (long long)SS * SS, (long long)SS * HH, (long long)SS * HH, 1.0f, 0);

        layernorm_kernel<<<MM, blk, 0, stream>>>(t0_f, g1 + (size_t)l * HH,
                                                 be1 + (size_t)l * HH, o1_f, o1_b);

        // FFN: h1 = relu(o1 @ W1 + b1);  t0 = h1 @ W2 + b2 + o1
        gemm(o1_b, w1, b1l, nullptr, nullptr, h1_b, MM, HH, HH, 1, 0, 0, 0,
             1.0f, GF_BF32 | GF_RELU);
        gemm(h1_b, w2, b2l, o1_f, t0_f, nullptr, MM, HH, HH, 1, 0, 0, 0,
             1.0f, GF_BF32);

        layernorm_kernel<<<MM, blk, 0, stream>>>(t0_f, g2 + (size_t)l * HH,
                                                 be2 + (size_t)l * HH, z_f, z_b);
    }

    // zf = z @ fc_w + fc_b
    gemm(z_b, fcw, fcb, nullptr, nullptr, zf_b, MM, HH, HH, 1, 0, 0, 0, 1.0f, GF_BF32);

    // logits = zf @ emb^T  (emb is [V,H]: K-contiguous per column -> GF_BT path)
    gemm(zf_b, emb, nullptr, nullptr, out, nullptr, MM, VV, HH, 1, 0, 0, 0,
         1.0f, GF_BT | GF_BF32);

    softmax_v_kernel<<<MM, blk, 0, stream>>>(out);
}
